// MathematicalAttention_86139864089199
// MI455X (gfx1250) — matrix operand order repaired and checked by reading, not compiled, not run on hardware
//
#include <hip/hip_runtime.h>
#include <hip/hip_bf16.h>
#include <stdint.h>

// ---------------- CDNA5 (gfx1250) types ----------------
typedef __attribute__((ext_vector_type(16))) __bf16 bf16x16;
typedef __attribute__((ext_vector_type(8)))  float  floatx8;

union Frag {
    bf16x16  v;
    uint32_t d[8];
    uint4    q[2];
};

#define WMMA_BF16(A, B, C) \
    __builtin_amdgcn_wmma_f32_16x16x32_bf16(false, (A), false, (B), (short)0, (C), false, false)

// Problem constants
#define BATCH 8
#define SEQ   1024
#define DMODEL 1024
#define NHEAD 16
#define HDIM  64

static __device__ __forceinline__ uint32_t f2bf(float f) {
    uint32_t u = __float_as_uint(f);
    return (u + 0x7FFFu + ((u >> 16) & 1u)) >> 16;  // round-to-nearest-even
}
static __device__ __forceinline__ uint32_t pack2bf(float lo, float hi) {
    return f2bf(lo) | (f2bf(hi) << 16);
}

// Async copy of one contiguous 2 KB tile (32 lanes x 16 B x 4 instructions)
// from global memory into LDS. Tracked by ASYNCcnt; the instruction offset is
// added to BOTH the LDS and global addresses (ISA 08_async_tensor §4.4), so a
// single address pair covers the whole tile.
static __device__ __forceinline__ void async_copy_2k(uint32_t lds, uint64_t gaddr) {
    asm volatile(
        "global_load_async_to_lds_b128 %0, %1, off\n\t"
        "global_load_async_to_lds_b128 %0, %1, off offset:512\n\t"
        "global_load_async_to_lds_b128 %0, %1, off offset:1024\n\t"
        "global_load_async_to_lds_b128 %0, %1, off offset:1536"
        :: "v"(lds), "v"(gaddr) : "memory");
}

// ---------------------------------------------------------------------------
// Kernel A: f32 -> bf16 conversion
// ---------------------------------------------------------------------------
__global__ void cvt_bf16_kernel(const float* __restrict__ in,
                                uint16_t* __restrict__ out, int n) {
    int i = blockIdx.x * blockDim.x + threadIdx.x;
    if (i < n) out[i] = (uint16_t)f2bf(in[i]);
}

// ---------------------------------------------------------------------------
// Kernel B: QKV projection.  Y[s,e] = sum_d X[s,d] * W[e,d] + b[e]
//   X bf16 [8192,1024] row-major; W bf16 [1024,1024] ([out,in] torch layout,
//   which is exactly N-major/K-contiguous for the WMMA B fragment).
//   z==0 -> Q [B,H,S,Dh] bf16 ; z==1 -> K [B,H,S,Dh] bf16 ;
//   z==2 -> V transposed [B,H,Dh,S] bf16 (so PV B-fragments are contiguous).
// Block = 128 threads (4 wave32); wave w owns a 16-row M tile; 64x64 C tile.
// ---------------------------------------------------------------------------
__global__ __launch_bounds__(128)
void qkv_proj_kernel(const uint16_t* __restrict__ Xb,
                     const uint16_t* __restrict__ Wqb,
                     const uint16_t* __restrict__ Wkb,
                     const uint16_t* __restrict__ Wvb,
                     const float* __restrict__ bq,
                     const float* __restrict__ bk,
                     const float* __restrict__ bv,
                     uint16_t* __restrict__ Qb,
                     uint16_t* __restrict__ Kb,
                     uint16_t* __restrict__ Vt) {
    const int z = blockIdx.z;
    const uint16_t* Wb  = (z == 0) ? Wqb : (z == 1) ? Wkb : Wvb;
    const float*    bia = (z == 0) ? bq  : (z == 1) ? bk  : bv;

    const int wave = threadIdx.x >> 5;
    const int lane = threadIdx.x & 31;
    const int l16  = lane & 15;      // A row / B column within tile
    const int hi   = lane >> 4;      // half-wave selector

    const int mbase = blockIdx.x * 64 + wave * 16;  // global s-row base
    const int ebase = blockIdx.y * 64;              // output-feature base

    floatx8 c[4] = {};

    for (int k = 0; k < DMODEL; k += 32) {
        // A fragment: 16x32 bf16 of X (row l16; K halves per documented layout)
        Frag a;
        const uint16_t* ap = Xb + (size_t)(mbase + l16) * DMODEL + k + hi * 8;
        a.q[0] = *(const uint4*)(ap);       // K = k0 .. k0+7
        a.q[1] = *(const uint4*)(ap + 16);  // K = k0+16 .. k0+23
#pragma unroll
        for (int nt = 0; nt < 4; ++nt) {
            // B fragment: 32x16 bf16 of W^T; W is [e][d] = N-major K-contig
            Frag b;
            const uint16_t* bp =
                Wb + (size_t)(ebase + nt * 16 + l16) * DMODEL + k + hi * 8;
            b.q[0] = *(const uint4*)(bp);
            b.q[1] = *(const uint4*)(bp + 16);
            c[nt] = WMMA_BF16(a.v, b.v, c[nt]);
        }
    }

    // Bias + scatter-store into attention-friendly layouts (bf16).
#pragma unroll
    for (int nt = 0; nt < 4; ++nt) {
        const int   e  = ebase + nt * 16 + l16;
        const float bb = bia[e];
        const int   h  = e >> 6;
        const int   dh = e & 63;
#pragma unroll
        for (int r = 0; r < 8; ++r) {
            const int sg = mbase + r + hi * 8;      // global row in [0,8192)
            const int b  = sg >> 10;
            const int s  = sg & 1023;
            const uint16_t yb = (uint16_t)f2bf(c[nt][r] + bb);
            if (z == 2) {
                Vt[(((size_t)(b * NHEAD + h) * HDIM + dh) << 10) + s] = yb;
            } else {
                uint16_t* O = (z == 0) ? Qb : Kb;
                O[(((size_t)(b * NHEAD + h) << 10) + s) * HDIM + dh] = yb;
            }
        }
    }
}

// ---------------------------------------------------------------------------
// Kernel C: fused  scores = Q K^T / 8 + bias[h] -> softmax -> attn out,
//           context = P V.  One wave32 per (b, h, 16-row q tile).
//   LDS: 16 x 1024 f32 probability tile (64 KB) + 2 x 2 KB K staging buffers
//   filled by GLOBAL_LOAD_ASYNC_TO_LDS_B128 (double-buffered on ASYNCcnt).
// ---------------------------------------------------------------------------
__global__ __launch_bounds__(32)
void fused_attn_kernel(const uint16_t* __restrict__ Qb,
                       const uint16_t* __restrict__ Kb,
                       const uint16_t* __restrict__ Vt,
                       const float* __restrict__ sbias,
                       float* __restrict__ ctx,
                       float* __restrict__ attn) {
    __shared__ float    P[16 * SEQ];          // 64 KB probability tile
    __shared__ uint16_t Kst[2][16 * HDIM];    // 2 x 2 KB async K staging

    const int lane  = threadIdx.x & 31;
    const int l16   = lane & 15;
    const int hi    = lane >> 4;
    const int qbase = blockIdx.x * 16;
    const int h     = blockIdx.y;
    const int b     = blockIdx.z;

    const uint16_t* Qh = Qb + (size_t)(b * NHEAD + h) * SEQ * HDIM;   // [S][64]
    const uint16_t* Kh = Kb + (size_t)(b * NHEAD + h) * SEQ * HDIM;   // [S][64]
    const uint16_t* Vh = Vt + (size_t)(b * NHEAD + h) * HDIM * SEQ;   // [64][S]
    const float sb = sbias[h];
    const float scale = 0.125f;  // 1/sqrt(64)

    // Warm caches for the V^T slice consumed in phase 3 (global_prefetch_b8).
    __builtin_prefetch(Vh + (size_t)lane * SEQ, 0, 3);
    __builtin_prefetch(Vh + (size_t)(lane + 32) * SEQ, 0, 3);

    // Hoisted Q A-fragments for K-dim (=Dh) steps k=0 and k=32.
    Frag a0, a1;
    {
        const uint16_t* ap = Qh + (size_t)(qbase + l16) * HDIM + hi * 8;
        a0.q[0] = *(const uint4*)(ap);
        a0.q[1] = *(const uint4*)(ap + 16);
        a1.q[0] = *(const uint4*)(ap + 32);
        a1.q[1] = *(const uint4*)(ap + 48);
    }

    // ---- Phase 1: scores -> LDS, K tiles staged via async-to-LDS ----------
    // Each 16x64 bf16 K tile is contiguous 2048 B in [S][Dh] layout.
    const uint32_t ldsK0 = (uint32_t)(uintptr_t)&Kst[0][0] + lane * 16;
    const uint32_t ldsK1 = (uint32_t)(uintptr_t)&Kst[1][0] + lane * 16;
    const uint64_t gK    = (uint64_t)(uintptr_t)Kh + (uint32_t)lane * 16;

    async_copy_2k(ldsK0, gK);  // prologue: tile 0 -> buffer 0
    for (int nt = 0; nt < 64; ++nt) {
        if (nt < 63) {
            async_copy_2k(((nt + 1) & 1) ? ldsK1 : ldsK0,
                          gK + (uint64_t)(nt + 1) * 2048u);
            // 8 outstanding; oldest 4 (tile nt) done when ASYNCcnt <= 4
            asm volatile("s_wait_asynccnt 0x4" ::: "memory");
        } else {
            asm volatile("s_wait_asynccnt 0x0" ::: "memory");
        }

        const uint16_t* bp = &Kst[nt & 1][l16 * HDIM + hi * 8];
        Frag b0, b1;  // K^T fragments: K rows are the WMMA N dim, Dh is K dim
        b0.q[0] = *(const uint4*)(bp);
        b0.q[1] = *(const uint4*)(bp + 16);
        b1.q[0] = *(const uint4*)(bp + 32);
        b1.q[1] = *(const uint4*)(bp + 48);

        floatx8 c = {};
        c = WMMA_BF16(a0.v, b0.v, c);
        c = WMMA_BF16(a1.v, b1.v, c);
#pragma unroll
        for (int r = 0; r < 8; ++r)
            P[(r + hi * 8) * SEQ + nt * 16 + l16] = c[r] * scale + sb;
    }
    __syncthreads();  // single-wave WG: LDS ops are in-order per wave

    // ---- Phase 2: softmax per row, write attention output, keep P in LDS --
    for (int r = 0; r < 16; ++r) {
        float* row = P + r * SEQ;
        float vals[32];
        float mx = -3.402823466e+38f;
#pragma unroll
        for (int j = 0; j < 32; ++j) {
            vals[j] = row[lane + j * 32];
            mx = fmaxf(mx, vals[j]);
        }
#pragma unroll
        for (int o = 16; o > 0; o >>= 1) mx = fmaxf(mx, __shfl_xor(mx, o, 32));
        float sum = 0.f;
#pragma unroll
        for (int j = 0; j < 32; ++j) {
            vals[j] = __expf(vals[j] - mx);
            sum += vals[j];
        }
#pragma unroll
        for (int o = 16; o > 0; o >>= 1) sum += __shfl_xor(sum, o, 32);
        const float inv = 1.f / sum;

        float* arow =
            attn + (((size_t)(b * NHEAD + h) * SEQ + (qbase + r)) * SEQ);
#pragma unroll
        for (int j = 0; j < 32; ++j) {
            const float p = vals[j] * inv;
            row[lane + j * 32]  = p;  // keep for PV
            arow[lane + j * 32] = p;  // HBM: written exactly once
        }
    }
    __syncthreads();

    // ---- Phase 3: context = P @ V  (A from LDS f32 -> bf16, B from Vt) ----
    floatx8 cc[4] = {};
    for (int k = 0; k < SEQ; k += 32) {
        Frag a;  // pack 16 probabilities into the 16x32 bf16 A layout
        const float* pr = P + l16 * SEQ + k + hi * 8;
        const float4 f0 = *(const float4*)(pr);
        const float4 f1 = *(const float4*)(pr + 4);
        const float4 f2 = *(const float4*)(pr + 16);
        const float4 f3 = *(const float4*)(pr + 20);
        a.d[0] = pack2bf(f0.x, f0.y);
        a.d[1] = pack2bf(f0.z, f0.w);
        a.d[2] = pack2bf(f1.x, f1.y);
        a.d[3] = pack2bf(f1.z, f1.w);
        a.d[4] = pack2bf(f2.x, f2.y);
        a.d[5] = pack2bf(f2.z, f2.w);
        a.d[6] = pack2bf(f3.x, f3.y);
        a.d[7] = pack2bf(f3.z, f3.w);
#pragma unroll
        for (int nt = 0; nt < 4; ++nt) {
            Frag bv;  // V^T rows are Dh (N dim), contiguous in S (K dim)
            const uint16_t* vp =
                Vh + (size_t)(nt * 16 + l16) * SEQ + k + hi * 8;
            bv.q[0] = *(const uint4*)(vp);
            bv.q[1] = *(const uint4*)(vp + 16);
            cc[nt] = WMMA_BF16(a.v, bv.v, cc[nt]);
        }
    }

    // Store context [B,S,D] f32
#pragma unroll
    for (int nt = 0; nt < 4; ++nt) {
#pragma unroll
        for (int r = 0; r < 8; ++r) {
            const int q   = qbase + r + hi * 8;
            const int col = h * HDIM + nt * 16 + l16;
            ctx[((size_t)b * SEQ + q) * DMODEL + col] = cc[nt][r];
        }
    }
}

// ---------------------------------------------------------------------------
extern "C" void kernel_launch(void* const* d_in, const int* in_sizes, int n_in,
                              void* d_out, int out_size, void* d_ws,
                              size_t ws_size, hipStream_t stream) {
    (void)in_sizes; (void)n_in; (void)out_size; (void)ws_size;

    const float* x   = (const float*)d_in[0];
    const float* Wq  = (const float*)d_in[1];
    const float* bq  = (const float*)d_in[2];
    const float* Wk  = (const float*)d_in[3];
    const float* bk  = (const float*)d_in[4];
    const float* Wv  = (const float*)d_in[5];
    const float* bv  = (const float*)d_in[6];
    const float* sb  = (const float*)d_in[7];

    float* ctx  = (float*)d_out;                                  // [8,1024,1024]
    float* attn = ctx + (size_t)BATCH * SEQ * DMODEL;             // [8,16,1024,1024]

    char* ws = (char*)d_ws;
    const size_t XB_BYTES = (size_t)BATCH * SEQ * DMODEL * 2;     // 16 MiB
    const size_t W_BYTES  = (size_t)DMODEL * DMODEL * 2;          // 2 MiB
    uint16_t* xb  = (uint16_t*)(ws);
    uint16_t* wqb = (uint16_t*)(ws + XB_BYTES);
    uint16_t* wkb = (uint16_t*)(ws + XB_BYTES + W_BYTES);
    uint16_t* wvb = (uint16_t*)(ws + XB_BYTES + 2 * W_BYTES);
    uint16_t* Qb  = (uint16_t*)(ws + XB_BYTES + 3 * W_BYTES);
    uint16_t* Kb  = (uint16_t*)((char*)Qb + XB_BYTES);
    uint16_t* Vt  = (uint16_t*)((char*)Kb + XB_BYTES);

    const int NX = BATCH * SEQ * DMODEL;   // 8388608
    const int NW = DMODEL * DMODEL;        // 1048576
    cvt_bf16_kernel<<<(NX + 255) / 256, 256, 0, stream>>>(x,  xb,  NX);
    cvt_bf16_kernel<<<(NW + 255) / 256, 256, 0, stream>>>(Wq, wqb, NW);
    cvt_bf16_kernel<<<(NW + 255) / 256, 256, 0, stream>>>(Wk, wkb, NW);
    cvt_bf16_kernel<<<(NW + 255) / 256, 256, 0, stream>>>(Wv, wvb, NW);

    qkv_proj_kernel<<<dim3(BATCH * SEQ / 64, DMODEL / 64, 3), 128, 0, stream>>>(
        xb, wqb, wkb, wvb, bq, bk, bv, Qb, Kb, Vt);

    fused_attn_kernel<<<dim3(SEQ / 16, NHEAD, BATCH), 32, 0, stream>>>(
        Qb, Kb, Vt, sb, ctx, attn);
}
